// Net_26345329393878
// MI455X (gfx1250) — compile-verified, matrix operand order checked by reading, not run on hardware
//
#include <hip/hip_runtime.h>
#include <hip/hip_bf16.h>
#include <stdint.h>

typedef __attribute__((ext_vector_type(16))) _Float16 v16h;
typedef __attribute__((ext_vector_type(8)))  float    v8f;

#define HID 128
#define HID2 256
#define EPSM 1e-7f

// ---------------------------------------------------------------- utilities
__global__ void zero_kernel(float* __restrict__ p, long n) {
    long i = (long)blockIdx.x * blockDim.x + threadIdx.x;
    long stride = (long)gridDim.x * blockDim.x;
    for (; i < n; i += stride) p[i] = 0.0f;
}

// ---------------------------------------------------------------- node encoder: h = x@wn + bn
__global__ void node_enc_kernel(const float* __restrict__ x, const float* __restrict__ wn,
                                const float* __restrict__ bn, float* __restrict__ h, int n) {
    int idx = blockIdx.x * blockDim.x + threadIdx.x;
    if (idx >= n * HID) return;
    int node = idx >> 7, c = idx & (HID - 1);
    float acc = bn[c];
#pragma unroll
    for (int f = 0; f < 6; ++f) acc += x[node * 6 + f] * wn[f * HID + c];
    h[idx] = acc;
}

// ---------------------------------------------------------------- fused message m = relu(h[src] + ea@we + be) + eps
__device__ __forceinline__ void compute_msg4(const float* __restrict__ h,
                                             const float* __restrict__ ea,
                                             const float* __restrict__ we,
                                             const float* __restrict__ be,
                                             int s, int e, int c, float m[4]) {
    float4 hv = *(const float4*)(h + (long)s * HID + c);
    float4 bv = *(const float4*)(be + c);
    float4 w0 = *(const float4*)(we + 0 * HID + c);
    float4 w1 = *(const float4*)(we + 1 * HID + c);
    float4 w2 = *(const float4*)(we + 2 * HID + c);
    float4 w3 = *(const float4*)(we + 3 * HID + c);
    float e0 = ea[e * 4 + 0], e1 = ea[e * 4 + 1], e2 = ea[e * 4 + 2], e3 = ea[e * 4 + 3];
    m[0] = fmaxf(hv.x + bv.x + e0 * w0.x + e1 * w1.x + e2 * w2.x + e3 * w3.x, 0.0f) + EPSM;
    m[1] = fmaxf(hv.y + bv.y + e0 * w0.y + e1 * w1.y + e2 * w2.y + e3 * w3.y, 0.0f) + EPSM;
    m[2] = fmaxf(hv.z + bv.z + e0 * w0.z + e1 * w1.z + e2 * w2.z + e3 * w3.z, 0.0f) + EPSM;
    m[3] = fmaxf(hv.w + bv.w + e0 * w0.w + e1 * w1.w + e2 * w2.w + e3 * w3.w, 0.0f) + EPSM;
}

// pass 1: zmax[dst] = segment_max(m)   (m > 0, so uint-punned atomicMax is correct)
__global__ void msg_max_kernel(const float* __restrict__ h, const float* __restrict__ ea,
                               const float* __restrict__ we, const float* __restrict__ be,
                               const int* __restrict__ src, const int* __restrict__ dst,
                               float* __restrict__ zmax, int nedges) {
    int e = blockIdx.x * 8 + (threadIdx.x >> 5);
    if (e >= nedges) return;
    int lane = threadIdx.x & 31;
    int c = lane * 4;
    int s = src[e], d = dst[e];
    float m[4];
    compute_msg4(h, ea, we, be, s, e, c, m);
    unsigned* zm = (unsigned*)(zmax + (long)d * HID + c);
#pragma unroll
    for (int j = 0; j < 4; ++j) atomicMax(&zm[j], __float_as_uint(m[j]));
}

// pass 2: denom[dst] += exp(m - zmax), numer[dst] += m*exp(m - zmax)
__global__ void msg_sum_kernel(const float* __restrict__ h, const float* __restrict__ ea,
                               const float* __restrict__ we, const float* __restrict__ be,
                               const int* __restrict__ src, const int* __restrict__ dst,
                               const float* __restrict__ zmax, float* __restrict__ denom,
                               float* __restrict__ numer, int nedges) {
    int e = blockIdx.x * 8 + (threadIdx.x >> 5);
    if (e >= nedges) return;
    int lane = threadIdx.x & 31;
    int c = lane * 4;
    int s = src[e], d = dst[e];
    float m[4];
    compute_msg4(h, ea, we, be, s, e, c, m);
    float4 zv = *(const float4*)(zmax + (long)d * HID + c);
    float z[4] = {zv.x, zv.y, zv.z, zv.w};
#pragma unroll
    for (int j = 0; j < 4; ++j) {
        float ez = __expf(m[j] - z[j]);
        atomicAdd(denom + (long)d * HID + c + j, ez);
        atomicAdd(numer + (long)d * HID + c + j, m[j] * ez);
    }
}

// hin = numer/(denom+1e-16) + h   (softmax-weighted aggregate + residual)
__global__ void combine_kernel(const float* __restrict__ numer, const float* __restrict__ denom,
                               const float* __restrict__ h, float* __restrict__ hin, long n) {
    long i = (long)blockIdx.x * blockDim.x + threadIdx.x;
    if (i >= n) return;
    hin[i] = numer[i] / (denom[i] + 1e-16f) + h[i];
}

// ---------------------------------------------------------------- weight repack: f32 [K,Ncols] -> f16 WMMA B-fragments
// Packed index: ((tc*KC + k0c)*32 + lane)*16 + i, where
//   col = tc*16 + (lane&15); kb = k0c*32 + (lane>>4)*8; k = kb + i + (i>=8 ? 8 : 0)
// so each lane's 16 halves of a B fragment are contiguous (32B) in memory.
__global__ void pack_w_kernel(const float* __restrict__ W, _Float16* __restrict__ Wp,
                              int K, int Ncols) {
    int idx = blockIdx.x * blockDim.x + threadIdx.x;
    if (idx >= K * Ncols) return;
    int i    = idx & 15;
    int lane = (idx >> 4) & 31;
    int kcix = idx >> 9;             // tc*KC + k0c
    int KC   = K >> 5;
    int tc   = kcix / KC;
    int k0c  = kcix % KC;
    int col  = tc * 16 + (lane & 15);
    int kb   = k0c * 32 + ((lane >> 4) << 3);
    int k    = kb + i + ((i >= 8) ? 8 : 0);
    Wp[idx] = (_Float16)W[(long)k * Ncols + col];
}

// ---------------------------------------------------------------- WMMA GEMM1: h1 = hin[N,128] @ w1[128,256] + b1
__global__ void gemm1_wmma_kernel(const float* __restrict__ A, const _Float16* __restrict__ Wp,
                                  const float* __restrict__ bias, float* __restrict__ C) {
    const int lane = threadIdx.x & 31;
    const int wid  = threadIdx.x >> 5;
    const int tr = blockIdx.x;              // row tile (N/16)
    const int tc = blockIdx.y * 4 + wid;    // col tile (256/16 = 16)
    const int row = tr * 16 + (lane & 15);
    const int col = tc * 16 + (lane & 15);
    const int ksub = (lane >> 4) * 8;
    const v16h* Bp = (const v16h*)Wp;
    const int KC = HID / 32;

    v8f acc = {};
#pragma unroll
    for (int k0 = 0; k0 < HID; k0 += 32) {
        const int kb = k0 + ksub;
        v16h a;
#pragma unroll
        for (int i = 0; i < 8; ++i) {
            a[i]     = (_Float16)A[(long)row * HID + kb + i];
            a[i + 8] = (_Float16)A[(long)row * HID + kb + 16 + i];
        }
        v16h b = Bp[(tc * KC + (k0 >> 5)) * 32 + lane];
        acc = __builtin_amdgcn_wmma_f32_16x16x32_f16(false, a, false, b, (short)0, acc,
                                                     false, false);
    }
    const float bv = bias[col];
    const int r0 = tr * 16 + (lane >> 4) * 8;
#pragma unroll
    for (int j = 0; j < 8; ++j) C[(long)(r0 + j) * HID2 + col] = acc[j] + bv;
}

// ---------------------------------------------------------------- batchnorm column stats
__global__ void colstats_kernel(const float* __restrict__ H1, float* __restrict__ sums, int nrows) {
    const int c = threadIdx.x;               // 256 channels
    int r0 = blockIdx.x * 128;
    int r1 = min(r0 + 128, nrows);
    float s = 0.0f, s2 = 0.0f;
    for (int r = r0; r < r1; ++r) {
        float v = H1[(long)r * HID2 + c];
        s += v; s2 += v * v;
    }
    atomicAdd(&sums[c], s);
    atomicAdd(&sums[HID2 + c], s2);
}

__global__ void finalize_stats_kernel(const float* __restrict__ sums, float* __restrict__ mean,
                                      float* __restrict__ rstd, float invn) {
    int c = threadIdx.x;                     // 256
    float mu = sums[c] * invn;
    float var = sums[HID2 + c] * invn - mu * mu;
    mean[c] = mu;
    rstd[c] = rsqrtf(var + 1e-5f);
}

// ---------------------------------------------------------------- WMMA GEMM2: h = relu( relu(BN(h1)) @ w2[256,128] + b2 )
__global__ void gemm2_wmma_kernel(const float* __restrict__ H1, const float* __restrict__ mean,
                                  const float* __restrict__ rstd, const float* __restrict__ gamma,
                                  const float* __restrict__ beta, const _Float16* __restrict__ Wp,
                                  const float* __restrict__ bias, float* __restrict__ Hout) {
    const int lane = threadIdx.x & 31;
    const int wid  = threadIdx.x >> 5;
    const int tr = blockIdx.x;              // row tile (N/16)
    const int tc = blockIdx.y * 4 + wid;    // col tile (128/16 = 8)
    const int row = tr * 16 + (lane & 15);
    const int col = tc * 16 + (lane & 15);
    const int ksub = (lane >> 4) * 8;
    const v16h* Bp = (const v16h*)Wp;
    const int KC = HID2 / 32;

    v8f acc = {};
#pragma unroll
    for (int k0 = 0; k0 < HID2; k0 += 32) {
        const int kb = k0 + ksub;
        v16h a;
#pragma unroll
        for (int i = 0; i < 8; ++i) {
            int ka = kb + i;
            int kc = kb + 16 + i;
            float va = (H1[(long)row * HID2 + ka] - mean[ka]) * rstd[ka] * gamma[ka] + beta[ka];
            float vc = (H1[(long)row * HID2 + kc] - mean[kc]) * rstd[kc] * gamma[kc] + beta[kc];
            a[i]     = (_Float16)fmaxf(va, 0.0f);
            a[i + 8] = (_Float16)fmaxf(vc, 0.0f);
        }
        v16h b = Bp[(tc * KC + (k0 >> 5)) * 32 + lane];
        acc = __builtin_amdgcn_wmma_f32_16x16x32_f16(false, a, false, b, (short)0, acc,
                                                     false, false);
    }
    const float bv = bias[col];
    const int r0 = tr * 16 + (lane >> 4) * 8;
#pragma unroll
    for (int j = 0; j < 8; ++j)
        Hout[(long)(r0 + j) * HID + col] = fmaxf(acc[j] + bv, 0.0f);
}

// ---------------------------------------------------------------- global mean pool + head
__global__ void pool_kernel(const float* __restrict__ h, const int* __restrict__ batch,
                            float* __restrict__ pooled, float* __restrict__ cnt) {
    int node = blockIdx.x;
    int c = threadIdx.x;                     // 128
    int b = batch[node];
    atomicAdd(&pooled[(long)b * HID + c], h[(long)node * HID + c]);
    if (c == 0) atomicAdd(&cnt[b], 1.0f);
}

__global__ void head_kernel(const float* __restrict__ pooled, const float* __restrict__ cnt,
                            const float* __restrict__ wd, const float* __restrict__ bd,
                            float* __restrict__ out, int ngraphs) {
    int g = blockIdx.x * (blockDim.x >> 5) + (threadIdx.x >> 5);
    int lane = threadIdx.x & 31;
    if (g >= ngraphs) return;
    float inv = 1.0f / fmaxf(cnt[g], 1.0f);
    float s = 0.0f;
#pragma unroll
    for (int c = lane; c < HID; c += 32) s += pooled[(long)g * HID + c] * wd[c];
#pragma unroll
    for (int off = 16; off > 0; off >>= 1) s += __shfl_xor(s, off, 32);
    if (lane == 0) out[g] = 1.0f / (1.0f + __expf(-(s * inv + bd[0])));
}

// ---------------------------------------------------------------- launch
extern "C" void kernel_launch(void* const* d_in, const int* in_sizes, int n_in,
                              void* d_out, int out_size, void* d_ws, size_t ws_size,
                              hipStream_t stream) {
    const float* x    = (const float*)d_in[0];
    const float* ea   = (const float*)d_in[1];
    const int*   eidx = (const int*)d_in[2];
    const int*   batch= (const int*)d_in[3];
    const float* wn   = (const float*)d_in[4];
    const float* bn   = (const float*)d_in[5];
    const float* we   = (const float*)d_in[6];
    const float* be   = (const float*)d_in[7];
    const float* cw1  = (const float*)d_in[8];
    const float* cb1  = (const float*)d_in[9];
    const float* cg   = (const float*)d_in[10];
    const float* cbt  = (const float*)d_in[11];
    const float* cw2  = (const float*)d_in[12];
    const float* cb2  = (const float*)d_in[13];
    const float* wd   = (const float*)d_in[14];
    const float* bd   = (const float*)d_in[15];
    float* out = (float*)d_out;

    const int N = in_sizes[3];           // 50000 (multiple of 16)
    const int E = in_sizes[1] / 4;       // 600000
    const int G = out_size;              // 512

    // workspace layout (floats); zmax/denom/numer kept contiguous so one zero pass covers them
    float* ws     = (float*)d_ws;
    float* h      = ws;                        // N*128
    float* hin    = h      + (long)N * HID;    // N*128
    float* zmax   = hin    + (long)N * HID;    // N*128 (contiguous block start)
    float* denom  = zmax   + (long)N * HID;    // N*128 (contiguous)
    float* numer  = denom  + (long)N * HID;    // N*128 (contiguous block end)
    float* h1     = numer  + (long)N * HID;    // N*256
    float* sums   = h1     + (long)N * HID2;   // 512
    float* mean   = sums   + 2 * HID2;         // 256
    float* rstd   = mean   + HID2;             // 256
    float* pooled = rstd   + HID2;             // G*128
    float* cnt    = pooled + (long)G * HID;    // G
    uintptr_t pw  = (uintptr_t)(cnt + G);
    pw = (pw + 31) & ~(uintptr_t)31;           // 32B align for v16h fragment loads
    _Float16* wpack1 = (_Float16*)pw;          // 128*256 halves (64KB)
    _Float16* wpack2 = wpack1 + HID * HID2;    // 256*128 halves (64KB)

    const int* srcI = eidx;
    const int* dstI = eidx + E;

    // node encoder
    node_enc_kernel<<<(N * HID + 255) / 256, 256, 0, stream>>>(x, wn, bn, h, N);

    const dim3 g1grid(N / 16, HID2 / 16 / 4);  // 16 col tiles, 4 waves/block
    const dim3 g2grid(N / 16, HID  / 16 / 4);  // 8 col tiles,  4 waves/block
    const int egrid = (E + 7) / 8;             // 8 waves/block, 1 wave/edge
    const int pgrid = (HID * HID2 + 255) / 256;

    for (int layer = 0; layer < 3; ++layer) {
        const float* w1 = cw1 + (long)layer * HID * HID2;
        const float* b1 = cb1 + (long)layer * HID2;
        const float* gg = cg  + (long)layer * HID2;
        const float* bt = cbt + (long)layer * HID2;
        const float* w2 = cw2 + (long)layer * HID2 * HID;
        const float* b2 = cb2 + (long)layer * HID;

        // zero zmax/denom/numer (contiguous) + BN stat accumulators
        zero_kernel<<<2048, 256, 0, stream>>>(zmax, 3L * N * HID);
        zero_kernel<<<2, 256, 0, stream>>>(sums, 2 * HID2);

        // repack this layer's weights into f16 WMMA fragment order
        pack_w_kernel<<<pgrid, 256, 0, stream>>>(w1, wpack1, HID, HID2);
        pack_w_kernel<<<pgrid, 256, 0, stream>>>(w2, wpack2, HID2, HID);

        // scatter-softmax message passing
        msg_max_kernel<<<egrid, 256, 0, stream>>>(h, ea, we, be, srcI, dstI, zmax, E);
        msg_sum_kernel<<<egrid, 256, 0, stream>>>(h, ea, we, be, srcI, dstI, zmax,
                                                  denom, numer, E);
        combine_kernel<<<(int)(((long)N * HID + 255) / 256), 256, 0, stream>>>(
            numer, denom, h, hin, (long)N * HID);

        // MLP with WMMA (fp16 in, fp32 accumulate)
        gemm1_wmma_kernel<<<g1grid, 128, 0, stream>>>(hin, wpack1, b1, h1);
        colstats_kernel<<<(N + 127) / 128, 256, 0, stream>>>(h1, sums, N);
        finalize_stats_kernel<<<1, 256, 0, stream>>>(sums, mean, rstd, 1.0f / (float)N);
        gemm2_wmma_kernel<<<g2grid, 128, 0, stream>>>(h1, mean, rstd, gg, bt, w2 ? wpack2 : wpack2, b2, h);
    }

    // global mean pool + dense head + sigmoid
    zero_kernel<<<64, 256, 0, stream>>>(pooled, (long)G * HID + G);
    pool_kernel<<<N, 128, 0, stream>>>(h, batch, pooled, cnt);
    head_kernel<<<(G + 7) / 8, 256, 0, stream>>>(pooled, cnt, wd, bd, out, G);
}
